// FourierSSM_46883863003256
// MI455X (gfx1250) — matrix operand order repaired and checked by reading, not compiled, not run on hardware
//
#include <hip/hip_runtime.h>
#include <hip/hip_bf16.h>

#define B_   64
#define T_   2048
#define D_   64
#define E_   128
#define H_   128
#define R_   8
#define WIN_ 64
#define HOP_ 32
#define FN_  33
#define TP_  63

typedef __attribute__((ext_vector_type(16))) _Float16 v16h;
typedef __attribute__((ext_vector_type(8)))  _Float16 v8h;
typedef __attribute__((ext_vector_type(8)))  float    v8f;

__device__ __forceinline__ float2 cmul2(float2 a, float2 b) {
  return make_float2(a.x*b.x - a.y*b.y, a.x*b.y + a.y*b.x);
}
__device__ __forceinline__ float2 cadd2(float2 a, float2 b) {
  return make_float2(a.x + b.x, a.y + b.y);
}

// ---------------------------------------------------------------------------
// Generic tiled WMMA GEMM:  C[M,N] = A[M,K](row major, lda) @ W[N,K]^T (+bias)
// f32 inputs converted to f16 in LDS; f32 accumulation via v_wmma_f32_16x16x32_f16.
// Block tile 128x64, 256 threads = 8 waves in a 4(M) x 2(N) grid; each wave
// computes a 32x32 tile with 4 WMMAs per K-step (A/B fragments reused 2x each).
// Tile loads: uniform-branch fast path with float2 (8B-aligned) vector loads +
// next-step prefetch; guarded scalar path only for the K tail.
// cstride/coff allow writing interleaved complex planes. EPI==2 -> atomicAdd.
// ---------------------------------------------------------------------------
template <int EPI>
__global__ __launch_bounds__(256)
void gemm_f16wmma(const float* __restrict__ A, int lda,
                  const float* __restrict__ W, int ldw,
                  const float* __restrict__ bias,
                  float* __restrict__ C, int ldc, int cstride, int coff,
                  int M, int N, int K, int kchunk)
{
  // 40-half row stride = 80 bytes: multiple of 16B (aligned v8h accesses), conflict-padded.
  __shared__ __align__(16) _Float16 As[128][40];
  __shared__ __align__(16) _Float16 Bs[64][40];

  const int tid  = threadIdx.x;
  const int lane = tid & 31;
  const int wave = tid >> 5;
  const int wm   = wave & 3;   // 0..3 -> 32-row M subtile
  const int wn   = wave >> 2;  // 0..1 -> 32-col N subtile
  const int rowBlock = blockIdx.y * 128;
  const int colBlock = blockIdx.x * 64;
  const int ks = blockIdx.z * kchunk;
  const int ke = min(K, ks + kchunk);

  v8f acc[2][2] = {};

  const int lr = tid >> 2;         // 0..63
  const int lc = (tid & 3) * 8;    // 0,8,16,24

  for (int k0 = ks; k0 < ke; k0 += 32) {
    const bool full = (k0 + 32 <= K);   // uniform (scalar) branch
    __syncthreads();
    // ---- A tile: 128 rows x 32 k (two row-groups per thread) ----
    #pragma unroll
    for (int g = 0; g < 2; ++g) {
      const int r = lr + g * 64;
      const int grow = rowBlock + r;
      v8h v = {};
      if (grow < M) {
        const float* ap = A + (size_t)grow * lda + k0 + lc;
        if (full) {
          const float2* ap2 = (const float2*)ap;     // 8B aligned: lda even, k0%32==0, lc%8==0
          float2 f0 = ap2[0], f1 = ap2[1], f2 = ap2[2], f3 = ap2[3];
          v[0] = (_Float16)f0.x; v[1] = (_Float16)f0.y;
          v[2] = (_Float16)f1.x; v[3] = (_Float16)f1.y;
          v[4] = (_Float16)f2.x; v[5] = (_Float16)f2.y;
          v[6] = (_Float16)f3.x; v[7] = (_Float16)f3.y;
          __builtin_prefetch(ap + 32, 0, 1);         // next K-step
        } else {
          #pragma unroll
          for (int i = 0; i < 8; ++i) {
            int gk = k0 + lc + i;
            v[i] = (gk < K) ? (_Float16)ap[i] : (_Float16)0.0f;
          }
        }
      }
      *(v8h*)&As[r][lc] = v;
    }
    // ---- B tile: W is (N,K) row-major -> k-contiguous per n row ----
    {
      const int gcol = colBlock + lr;
      v8h v = {};
      if (gcol < N) {
        const float* wp = W + (size_t)gcol * ldw + k0 + lc;
        if (full) {
          const float2* wp2 = (const float2*)wp;
          float2 f0 = wp2[0], f1 = wp2[1], f2 = wp2[2], f3 = wp2[3];
          v[0] = (_Float16)f0.x; v[1] = (_Float16)f0.y;
          v[2] = (_Float16)f1.x; v[3] = (_Float16)f1.y;
          v[4] = (_Float16)f2.x; v[5] = (_Float16)f2.y;
          v[6] = (_Float16)f3.x; v[7] = (_Float16)f3.y;
          __builtin_prefetch(wp + 32, 0, 1);
        } else {
          #pragma unroll
          for (int i = 0; i < 8; ++i) {
            int gk = k0 + lc + i;
            v[i] = (gk < K) ? (_Float16)wp[i] : (_Float16)0.0f;
          }
        }
      }
      *(v8h*)&Bs[lr][lc] = v;
    }
    __syncthreads();

    // ---- fragments ----
    // A frag (16x32 f16): lane L holds row m=L%16; k in {b..b+7} u {b+16..b+23}, b=8*(L/16)
    const int hsel = lane >> 4;
    v16h af[2];
    #pragma unroll
    for (int i = 0; i < 2; ++i) {
      const v8h* ap = (const v8h*)&As[wm * 32 + i * 16 + (lane & 15)][0];
      af[i] = __builtin_shufflevector(ap[hsel], ap[hsel + 2],
                                      0,1,2,3,4,5,6,7,8,9,10,11,12,13,14,15);
    }
    // B frag (32x16 f16): lane L holds col n=L%16; k in {b..b+7} u {b+16..b+23}, b=8*(L/16)
    v16h bf[2];
    #pragma unroll
    for (int j = 0; j < 2; ++j) {
      const v8h* bp = (const v8h*)&Bs[wn * 32 + j * 16 + (lane & 15)][0];
      bf[j] = __builtin_shufflevector(bp[hsel], bp[hsel + 2],
                                      0,1,2,3,4,5,6,7,8,9,10,11,12,13,14,15);
    }
    #pragma unroll
    for (int i = 0; i < 2; ++i)
      #pragma unroll
      for (int j = 0; j < 2; ++j)
        acc[i][j] = __builtin_amdgcn_wmma_f32_16x16x32_f16(false, af[i], false, bf[j],
                                                           (short)0, acc[i][j], false, false);
  }

  // C/D layout: lane L -> N = L%16 (+subtile), VGPR r -> M = r + 8*(L/16)
  #pragma unroll
  for (int i = 0; i < 2; ++i) {
    #pragma unroll
    for (int r = 0; r < 8; ++r) {
      int grow = rowBlock + wm * 32 + i * 16 + (lane >> 4) * 8 + r;
      if (grow >= M) continue;
      #pragma unroll
      for (int j = 0; j < 2; ++j) {
        int gcol = colBlock + wn * 32 + j * 16 + (lane & 15);
        if (gcol >= N) continue;
        float v = acc[i][j][r];
        float* cp = C + (size_t)grow * ldc + (size_t)gcol * cstride + coff;
        if (EPI == 2) {
          atomicAdd(cp, v);
        } else {
          if (bias) v += bias[gcol];
          *cp = v;
        }
      }
    }
  }
}

// ---------------------------------------------------------------------------
// Constants: STFT DFT matrices (Hamming folded in), iDFT (window folded in),
// G = 1/(1 - A*exp(-2πi f/WIN)), overlap-add window-square sums.
// ---------------------------------------------------------------------------
__global__ void const_init(const float2* __restrict__ Acoef,
                           float* __restrict__ Wc, float* __restrict__ Ws,
                           float* __restrict__ Wi, float2* __restrict__ G,
                           float* __restrict__ wsum)
{
  const int tid = blockIdx.x * 256 + threadIdx.x;
  const float PI2 = 6.28318530717958647692f;
  if (tid < FN_ * WIN_) {              // Wc/Ws: (N=33, K=64)
    int f = tid / WIN_, k = tid % WIN_;
    float ham = 0.54f - 0.46f * __cosf(PI2 * k / WIN_);
    float th  = PI2 * f * k / WIN_;
    Wc[tid] =  ham * __cosf(th);
    Ws[tid] = -ham * __sinf(th);
  }
  if (tid < WIN_ * FN_) {              // Wi: (N=64 samples, K=66 re/im)
    int k = tid / FN_, f = tid % FN_;
    float cf  = (f == 0 || f == FN_ - 1) ? 1.0f : 2.0f;
    float ham = 0.54f - 0.46f * __cosf(PI2 * k / WIN_);
    float th  = PI2 * f * k / WIN_;
    Wi[k * (2 * FN_) + 2 * f]     =  cf * __cosf(th) * ham / WIN_;
    Wi[k * (2 * FN_) + 2 * f + 1] = -cf * __sinf(th) * ham / WIN_;
  }
  if (tid < H_ * FN_) {                // G[n,f]
    int n = tid / FN_, f = tid % FN_;
    float ph = PI2 * f / WIN_;
    float c = __cosf(ph), s = __sinf(ph);
    float2 a = Acoef[n];
    float dr = 1.0f - (a.x * c + a.y * s);
    float di = a.x * s - a.y * c;
    float inv = 1.0f / (dr * dr + di * di);
    G[tid] = make_float2(dr * inv, -di * inv);
  }
  if (tid < T_) {                      // overlap-add normalizer
    int n = tid, t1 = n >> 5;
    float acc = 0.0f;
    for (int t = t1 - 1; t <= t1; ++t) {
      if (t >= 0 && t < TP_) {
        int off = n - t * HOP_;
        if (off >= 0 && off < WIN_) {
          float ham = 0.54f - 0.46f * __cosf(PI2 * off / WIN_);
          acc += ham * ham;
        }
      }
    }
    wsum[n] = acc;
  }
}

// gather STFT frames: frames[(b,e,t),k] = xe[b, t*HOP+k, e]
__global__ void gather_frames(const float* __restrict__ xe, float* __restrict__ frames)
{
  long tid = (long)blockIdx.x * 256 + threadIdx.x;
  if (tid >= (long)B_ * TP_ * WIN_ * E_) return;
  int e = tid % E_;
  int k = (tid / E_) % WIN_;
  int t = (tid / ((long)E_ * WIN_)) % TP_;
  int b = tid / ((long)E_ * WIN_ * TP_);
  float v = xe[((size_t)b * T_ + t * HOP_ + k) * E_ + e];
  frames[(((size_t)b * E_ + e) * TP_ + t) * WIN_ + k] = v;
}

// Xcat[b,t, f*2E + e] = Re(spec[b,e,t,f]);  [... + E + e] = Im
__global__ void build_xcat(const float2* __restrict__ spec, float* __restrict__ Xcat)
{
  long tid = (long)blockIdx.x * 256 + threadIdx.x;
  if (tid >= (long)B_ * E_ * TP_ * FN_) return;
  int f = tid % FN_;
  int t = (tid / FN_) % TP_;
  int e = (tid / (FN_ * TP_)) % E_;
  int b = tid / ((long)FN_ * TP_ * E_);
  float2 v = spec[tid];
  size_t base = (((size_t)b * TP_ + t) * FN_ + f) * (2 * E_);
  Xcat[base + e]      = v.x;
  Xcat[base + E_ + e] = v.y;
}

// LayerNorm (+ optional SiLU) over last dim W, in place; bias already in X.
template <int W, bool SILU>
__global__ void ln_act(float* __restrict__ X, const float* __restrict__ g,
                       const float* __restrict__ b)
{
  __shared__ float red[W];
  const int row = blockIdx.x, t = threadIdx.x;
  float v = X[(size_t)row * W + t];
  red[t] = v; __syncthreads();
  #pragma unroll
  for (int s = W / 2; s > 0; s >>= 1) { if (t < s) red[t] += red[t + s]; __syncthreads(); }
  float mean = red[0] / W; __syncthreads();
  float d = v - mean;
  red[t] = d * d; __syncthreads();
  #pragma unroll
  for (int s = W / 2; s > 0; s >>= 1) { if (t < s) red[t] += red[t + s]; __syncthreads(); }
  float var = red[0] / W;
  float o = d * rsqrtf(var + 1e-5f) * g[t] + b[t];
  if (SILU) o = o / (1.0f + __expf(-o));
  X[(size_t)row * W + t] = o;
}

// ---------------------------------------------------------------------------
// Hd[b,t,e,f] = sum_n G[n,f] * (V_C[e,:]·Ub[b,t,n,:]) * (V_B[e,:]·Uc[b,t,n,:])
// One block per (b,t); H processed in chunks of 16 with dot-products staged in LDS.
// ---------------------------------------------------------------------------
__global__ __launch_bounds__(256)
void hd_kernel(const float* __restrict__ params, const float2* __restrict__ VB,
               const float2* __restrict__ VC, const float2* __restrict__ G,
               float2* __restrict__ Hd)
{
  const int p = blockIdx.x;                       // b*TP + t
  const float* pr = params + (size_t)p * 4096;    // [Br | Bi | Cr | Ci], each H*R
  __shared__ float2 sVB[E_][R_], sVC[E_][R_];
  __shared__ float2 prod[16][E_];
  __shared__ float2 sUb[16][R_], sUc[16][R_];
  __shared__ float2 sG[16][FN_];
  const int tid = threadIdx.x;

  for (int i = tid; i < E_ * R_; i += 256) {
    sVB[i >> 3][i & 7] = VB[i];
    sVC[i >> 3][i & 7] = VC[i];
  }
  float2 acc[17];
  #pragma unroll
  for (int i = 0; i < 17; ++i) acc[i] = make_float2(0.f, 0.f);

  for (int nc = 0; nc < H_ / 16; ++nc) {
    __syncthreads();
    for (int i = tid; i < 16 * R_; i += 256) {
      int n2 = i >> 3, r = i & 7;
      int col = (nc * 16 + n2) * R_ + r;
      sUb[n2][r] = make_float2(pr[col],        pr[1024 + col]);
      sUc[n2][r] = make_float2(pr[2048 + col], pr[3072 + col]);
    }
    for (int i = tid; i < 16 * FN_; i += 256) {
      int n2 = i / FN_, f = i % FN_;
      sG[n2][f] = G[(nc * 16 + n2) * FN_ + f];
    }
    __syncthreads();
    for (int i = tid; i < 16 * E_; i += 256) {
      int n2 = i >> 7, e = i & 127;
      float2 dc = make_float2(0.f, 0.f), db = make_float2(0.f, 0.f);
      #pragma unroll
      for (int r = 0; r < R_; ++r) {
        dc = cadd2(dc, cmul2(sVC[e][r], sUb[n2][r]));
        db = cadd2(db, cmul2(sVB[e][r], sUc[n2][r]));
      }
      prod[n2][e] = cmul2(dc, db);
    }
    __syncthreads();
    #pragma unroll
    for (int i = 0; i < 17; ++i) {
      int idx = tid + i * 256;
      if (idx < E_ * FN_) {
        int e = idx / FN_, f = idx % FN_;
        float2 a = acc[i];
        #pragma unroll 4
        for (int n2 = 0; n2 < 16; ++n2)
          a = cadd2(a, cmul2(sG[n2][f], prod[n2][e]));
        acc[i] = a;
      }
    }
  }
  #pragma unroll
  for (int i = 0; i < 17; ++i) {
    int idx = tid + i * 256;
    if (idx < E_ * FN_) {
      int e = idx / FN_, f = idx % FN_;
      Hd[((size_t)p * E_ + e) * FN_ + f] = acc[i];
    }
  }
}

// ---------------------------------------------------------------------------
// Fused: running average over t + Heff = Havg + alpha*Hd, then K=3 iterations of
// X = Heff*X + X + bias_k with complex ReLU. One thread per (b,e,f), in place on X.
// ---------------------------------------------------------------------------
__global__ void mix_kernel(float2* __restrict__ X, const float2* __restrict__ Hd,
                           const float* __restrict__ alpha, const float2* __restrict__ biases)
{
  int tid = blockIdx.x * 256 + threadIdx.x;
  if (tid >= B_ * E_ * FN_) return;
  int f = tid % FN_;
  int e = (tid / FN_) % E_;
  int b = tid / (FN_ * E_);
  float2 sum = make_float2(0.f, 0.f);
  for (int t = 0; t < TP_; ++t) {
    float2 hd = Hd[(((size_t)b * TP_ + t) * E_ + e) * FN_ + f];
    sum.x += hd.x; sum.y += hd.y;
    float inv = 1.0f / (float)(t + 1);
    float al  = alpha[t];
    float2 h = make_float2(sum.x * inv + al * hd.x, sum.y * inv + al * hd.y);
    size_t xi = (((size_t)b * E_ + e) * TP_ + t) * FN_ + f;
    float2 x = X[xi];
    #pragma unroll
    for (int k = 0; k < 3; ++k) {
      float2 bb = biases[(((size_t)k * TP_ + t) * E_ + e) * FN_ + f];
      float nr = h.x * x.x - h.y * x.y + x.x + bb.x;
      float ni = h.x * x.y + h.y * x.x + x.y + bb.y;
      x.x = fmaxf(nr, 0.f); x.y = fmaxf(ni, 0.f);
    }
    X[xi] = x;
  }
}

// overlap-add gather (each sample covered by <=2 frames) + /wsum; writes y[b,t,e]
__global__ void overlap_add(const float* __restrict__ frames, const float* __restrict__ wsum,
                            float* __restrict__ y)
{
  int tid = blockIdx.x * 256 + threadIdx.x;
  if (tid >= B_ * E_ * T_) return;
  int n  = tid & (T_ - 1);
  int be = tid >> 11;
  int t1 = n >> 5;
  float acc = 0.f;
  for (int t = t1 - 1; t <= t1; ++t) {
    if (t >= 0 && t < TP_) {
      int off = n - t * HOP_;
      if (off >= 0 && off < WIN_)
        acc += frames[((size_t)be * TP_ + t) * WIN_ + off];
    }
  }
  int b = be >> 7, e = be & 127;
  y[((size_t)b * T_ + n) * E_ + e] = acc / wsum[n];
}

__global__ void zero_f(float* __restrict__ p, int n)
{
  int i = blockIdx.x * 256 + threadIdx.x;
  if (i < n) p[i] = 0.f;
}

// leaky_relu(f1+bf1) -> dot Wf2 -> softplus
__global__ __launch_bounds__(128)
void head_kernel(const float* __restrict__ f1, const float* __restrict__ bf1,
                 const float* __restrict__ Wf2, const float* __restrict__ bf2,
                 float* __restrict__ f2)
{
  __shared__ float s[512];
  const int b = blockIdx.x;
  for (int j = threadIdx.x; j < 512; j += 128) {
    float v = f1[b * 512 + j] + bf1[j];
    s[j] = v > 0.f ? v : 0.01f * v;
  }
  __syncthreads();
  const int h = threadIdx.x;
  float acc = bf2[h];
  #pragma unroll 4
  for (int j = 0; j < 512; ++j) acc += s[j] * Wf2[h * 512 + j];
  float sp = (acc > 20.f) ? acc : log1pf(__expf(acc));
  f2[b * 128 + h] = sp;
}

// ---------------------------------------------------------------------------
extern "C" void kernel_launch(void* const* d_in, const int* in_sizes, int n_in,
                              void* d_out, int out_size, void* d_ws, size_t ws_size,
                              hipStream_t stream)
{
  const float*  x     = (const float*) d_in[0];
  const float*  Wemb  = (const float*) d_in[1];
  const float*  bemb  = (const float*) d_in[2];
  const float2* Acoef = (const float2*)d_in[3];
  const float2* V_B   = (const float2*)d_in[4];
  const float2* V_C   = (const float2*)d_in[5];
  const float*  alpha = (const float*) d_in[6];
  const float*  W1    = (const float*) d_in[7];
  const float*  b1    = (const float*) d_in[8];
  const float*  g1    = (const float*) d_in[9];
  const float*  be1   = (const float*) d_in[10];
  const float*  W2    = (const float*) d_in[11];
  const float*  g2    = (const float*) d_in[12];
  const float*  be2   = (const float*) d_in[13];
  const float*  W3    = (const float*) d_in[14];
  const float*  b3    = (const float*) d_in[15];
  const float2* biases= (const float2*)d_in[16];
  const float*  Winv  = (const float*) d_in[17];
  const float*  binv  = (const float*) d_in[18];
  const float*  Wf1   = (const float*) d_in[19];
  const float*  bf1   = (const float*) d_in[20];
  const float*  Wf2   = (const float*) d_in[21];
  const float*  bf2   = (const float*) d_in[22];
  const float*  Wf3   = (const float*) d_in[23];
  const float*  bf3   = (const float*) d_in[24];
  float* out = (float*)d_out;

  // ---- workspace carve ----
  char* w = (char*)d_ws;
  size_t off = 0;
  auto carve = [&](size_t bytes) -> void* {
    void* p = w + off;
    off += (bytes + 255) & ~(size_t)255;
    return p;
  };
  float*  xe     = (float*) carve((size_t)B_ * T_ * E_ * 4);            // 67MB; reused as y
  float*  reg1   = (float*) carve((size_t)B_ * TP_ * FN_ * 2 * E_ * 4); // 136MB; frames/Xcat/iframes
  float2* spec   = (float2*)carve((size_t)B_ * E_ * TP_ * FN_ * 8);     // 136MB; X in-place
  float*  h1     = (float*) carve((size_t)B_ * TP_ * 128 * 4);
  float*  h2     = (float*) carve((size_t)B_ * TP_ * 64 * 4);
  float*  params = (float*) carve((size_t)B_ * TP_ * 4096 * 4);         // 66MB
  float2* Hd     = (float2*)carve((size_t)B_ * TP_ * E_ * FN_ * 8);     // 136MB
  float*  y2     = (float*) carve((size_t)B_ * T_ * D_ * 4);            // 33.5MB
  float*  Wc     = (float*) carve(FN_ * WIN_ * 4);
  float*  Wsn    = (float*) carve(FN_ * WIN_ * 4);
  float*  Wi     = (float*) carve(WIN_ * 2 * FN_ * 4);
  float2* G      = (float2*)carve(H_ * FN_ * 8);
  float*  wsum   = (float*) carve(T_ * 4);
  float*  f1     = (float*) carve(64 * 512 * 4);
  float*  f2     = (float*) carve(64 * 128 * 4);
  (void)ws_size; (void)in_sizes; (void)n_in; (void)out_size;

  const int BT  = B_ * T_;          // 131072
  const int MST = B_ * E_ * TP_;    // 516096

  // 0) constants
  const_init<<<20, 256, 0, stream>>>(Acoef, Wc, Wsn, Wi, G, wsum);

  // 1) embed: xe[BT,128] = x[BT,64] @ Wemb^T + bemb
  gemm_f16wmma<0><<<dim3(2, BT / 128, 1), 256, 0, stream>>>(
      x, D_, Wemb, D_, bemb, xe, E_, 1, 0, BT, E_, D_, D_);

  // 2) gather windowless frames (window folded into DFT matrices)
  gather_frames<<<(B_ * TP_ * WIN_ * E_) / 256, 256, 0, stream>>>(xe, reg1);

  // 3) STFT = frames @ DFT: real and imag planes written interleaved into spec
  gemm_f16wmma<0><<<dim3(1, MST / 128, 1), 256, 0, stream>>>(
      reg1, WIN_, Wc, WIN_, nullptr, (float*)spec, 2 * FN_, 2, 0, MST, FN_, WIN_, WIN_);
  gemm_f16wmma<0><<<dim3(1, MST / 128, 1), 256, 0, stream>>>(
      reg1, WIN_, Wsn, WIN_, nullptr, (float*)spec, 2 * FN_, 2, 1, MST, FN_, WIN_, WIN_);

  // 4) Xcat (B*TP, 2*E*FN)
  build_xcat<<<(B_ * E_ * TP_ * FN_) / 256, 256, 0, stream>>>(spec, reg1);

  // 5) param_net
  const int PIN = 2 * E_ * FN_;  // 8448
  gemm_f16wmma<0><<<dim3(2, (B_ * TP_ + 127) / 128, 1), 256, 0, stream>>>(
      reg1, PIN, W1, PIN, b1, h1, 128, 1, 0, B_ * TP_, 128, PIN, PIN);
  ln_act<128, true><<<B_ * TP_, 128, 0, stream>>>(h1, g1, be1);
  gemm_f16wmma<0><<<dim3(1, (B_ * TP_ + 127) / 128, 1), 256, 0, stream>>>(
      h1, 128, W2, 128, nullptr, h2, 64, 1, 0, B_ * TP_, 64, 128, 128);
  ln_act<64, false><<<B_ * TP_, 64, 0, stream>>>(h2, g2, be2);
  gemm_f16wmma<0><<<dim3(64, (B_ * TP_ + 127) / 128, 1), 256, 0, stream>>>(
      h2, 64, W3, 64, b3, params, 4096, 1, 0, B_ * TP_, 4096, 64, 64);

  // 6) SSM projection diag(H_mix)
  hd_kernel<<<B_ * TP_, 256, 0, stream>>>(params, V_B, V_C, G, Hd);

  // 7) fused running-average + alpha + K=3 complex mixing/ReLU, in place on spec
  mix_kernel<<<(B_ * E_ * FN_ + 255) / 256, 256, 0, stream>>>(spec, Hd, alpha, biases);

  // 8) iSTFT: frames = specRI @ Wi^T (window folded in), then overlap-add -> y (reuse xe)
  gemm_f16wmma<0><<<dim3(1, MST / 128, 1), 256, 0, stream>>>(
      (const float*)spec, 2 * FN_, Wi, 2 * FN_, nullptr, reg1, WIN_, 1, 0,
      MST, WIN_, 2 * FN_, 2 * FN_);
  overlap_add<<<(B_ * E_ * T_) / 256, 256, 0, stream>>>(reg1, wsum, xe);

  // 9) Winv: y2[BT,64] = y[BT,128] @ Winv^T + binv
  gemm_f16wmma<0><<<dim3(1, BT / 128, 1), 256, 0, stream>>>(
      xe, E_, Winv, E_, binv, y2, D_, 1, 0, BT, D_, E_, E_);

  // 10) Wf1 (64 x 512 x 131072): split-K atomic accumulation
  zero_f<<<128, 256, 0, stream>>>(f1, 64 * 512);
  gemm_f16wmma<2><<<dim3(8, 1, 128), 256, 0, stream>>>(
      y2, T_ * D_, Wf1, T_ * D_, nullptr, f1, 512, 1, 0, 64, 512, T_ * D_, 1024);

  // 11) head: leaky_relu -> Wf2 -> softplus
  head_kernel<<<64, 128, 0, stream>>>(f1, bf1, Wf2, bf2, f2);

  // 12) final: out[64,12288] = f2 @ Wf3^T + bf3
  gemm_f16wmma<0><<<dim3(192, 1, 1), 256, 0, stream>>>(
      f2, 128, Wf3, 128, bf3, out, 12288, 1, 0, 64, 12288, 128, 128);
}
